// SoftAggONNX_v2_77730318123553
// MI455X (gfx1250) — matrix operand order repaired and checked by reading, not compiled, not run on hardware
//
#include <hip/hip_runtime.h>
#include <hip/hip_bf16.h>

// Problem constants (match the reference)
constexpr int Bq = 2;
constexpr int Nq = 32768;
constexpr int Cq = 512;
constexpr int Mq = 1024;

typedef __attribute__((ext_vector_type(16))) __bf16 v16bf;
typedef __attribute__((ext_vector_type(8)))  __bf16 v8bf;
typedef __attribute__((ext_vector_type(8)))  float  v8f;

union AB16 { v16bf v; v8bf h[2]; };

// ---- WMMA helpers -----------------------------------------------------------
__device__ __forceinline__ v8f wmma_bf16(v16bf a, v16bf b, v8f c) {
  // D = A(16x32 bf16) x B(32x16 bf16) + C(16x16 f32)
  return __builtin_amdgcn_wmma_f32_16x16x32_bf16(
      /*neg_a=*/false, a, /*neg_b=*/false, b,
      /*c_mod=*/(short)0, c, /*reuse_a=*/false, /*reuse_b=*/false);
}

// A fragment (16x32, bf16): lane l (l15 = l&15 -> row), hi = l>>4.
// lane holds K = [kk + hi*8 .. +7] and [kk + 16 + hi*8 .. +7]  (ISA 7.12.2)
__device__ __forceinline__ v16bf load_frag_a(const __bf16* __restrict__ rowptr,
                                             int kk, int hi) {
  AB16 f;
  f.h[0] = *reinterpret_cast<const v8bf*>(rowptr + kk + hi * 8);
  f.h[1] = *reinterpret_cast<const v8bf*>(rowptr + kk + 16 + hi * 8);
  return f.v;
}

// B fragment (32x16, bf16): lane n = l&15 holds column n, K = kk + (l>>4)*8 .. +7 and kk + 16 + (l>>4)*8 .. +7
// (two 8-bf16 runs of W[col, :], W row-major [d, c] => B = W^T)
__device__ __forceinline__ v16bf load_frag_b(const __bf16* __restrict__ colptr,
                                             int kk, int hi) {
  AB16 f;
  f.h[0] = *reinterpret_cast<const v8bf*>(colptr + kk + hi * 8);
  f.h[1] = *reinterpret_cast<const v8bf*>(colptr + kk + 16 + hi * 8);
  return f.v;
}

// A fragment from fp32 rows (pass 3: aggregated y is fp32), converted in-register
__device__ __forceinline__ v16bf load_frag_a_f32(const float* __restrict__ rowptr,
                                                 int kk, int hi) {
  AB16 f;
  const float* p0 = rowptr + kk + hi * 8;
  const float* p1 = rowptr + kk + 16 + hi * 8;
#pragma unroll
  for (int i = 0; i < 8; ++i) {
    f.v[i]     = (__bf16)p0[i];
    f.v[8 + i] = (__bf16)p1[i];
  }
  return f.v;
}

// ---- Utility kernels --------------------------------------------------------
__global__ void cvt_f32_to_bf16(const float* __restrict__ s,
                                __bf16* __restrict__ d, int n) {
  int i = blockIdx.x * blockDim.x + threadIdx.x;
  int stride = gridDim.x * blockDim.x;
  for (; i < n; i += stride) d[i] = (__bf16)s[i];
}

__global__ void zero_f32(float* __restrict__ p, int n) {
  int i = blockIdx.x * blockDim.x + threadIdx.x;
  int stride = gridDim.x * blockDim.x;
  for (; i < n; i += stride) p[i] = 0.0f;
}

// ---- Pass 1: logits = x*Wg^T + bg ; e = exp ; denom[b, ix, :] += e ----------
__global__ __launch_bounds__(256) void pass1_g_exp_scatter(
    const __bf16* __restrict__ xh, const __bf16* __restrict__ Wg,
    const float* __restrict__ bg, const int* __restrict__ ix,
    float* __restrict__ denom) {
  const int lane = threadIdx.x & 31;
  const int wave = threadIdx.x >> 5;
  const int l15 = lane & 15, hi = lane >> 4;
  const int r0 = (blockIdx.x * 8 + wave) * 16;  // 16-row tile per wave
  const __bf16* arow = xh + (size_t)(r0 + l15) * Cq;

  for (int nc = 0; nc < 8; ++nc) {  // 8 chunks of 4 n-tiles (64 cols)
    v8f acc[4] = {};
    for (int kk = 0; kk < Cq; kk += 32) {
      v16bf a = load_frag_a(arow, kk, hi);
      __builtin_prefetch(arow + kk + 32, 0, 1);  // global_prefetch_b8
#pragma unroll
      for (int t = 0; t < 4; ++t) {
        const int col = (nc * 4 + t) * 16 + l15;
        v16bf b = load_frag_b(Wg + (size_t)col * Cq, kk, hi);
        acc[t] = wmma_bf16(a, b, acc[t]);
      }
    }
#pragma unroll
    for (int t = 0; t < 4; ++t) {
      const int col = (nc * 4 + t) * 16 + l15;
      const float bias = bg[col];
#pragma unroll
      for (int j = 0; j < 8; ++j) {  // acc VGPR j: row = j + hi*8 (ISA layout)
        const int r = r0 + j + hi * 8;
        const int bb = r >> 15;           // N = 32768
        const int n = r & (Nq - 1);
        const int m = ix[n];
        const float e = __expf(acc[t][j] + bias);
        atomicAdd(denom + ((size_t)bb * Mq + m) * Cq + col, e);
      }
    }
  }
}

// ---- Pass 2: recompute g & f GEMMs (shared A frag), scatter fx * e/denom ----
__global__ __launch_bounds__(256) void pass2_fw_scatter(
    const __bf16* __restrict__ xh, const __bf16* __restrict__ Wg,
    const __bf16* __restrict__ Wf, const float* __restrict__ bg,
    const float* __restrict__ bf, const int* __restrict__ ix,
    const float* __restrict__ denom, float* __restrict__ yagg) {
  const int lane = threadIdx.x & 31;
  const int wave = threadIdx.x >> 5;
  const int l15 = lane & 15, hi = lane >> 4;
  const int r0 = (blockIdx.x * 8 + wave) * 16;
  const __bf16* arow = xh + (size_t)(r0 + l15) * Cq;

  for (int nc = 0; nc < 8; ++nc) {
    v8f accg[4] = {};
    v8f accf[4] = {};
    for (int kk = 0; kk < Cq; kk += 32) {
      v16bf a = load_frag_a(arow, kk, hi);  // loaded once, feeds 8 WMMAs
      __builtin_prefetch(arow + kk + 32, 0, 1);
#pragma unroll
      for (int t = 0; t < 4; ++t) {
        const int col = (nc * 4 + t) * 16 + l15;
        v16bf bgm = load_frag_b(Wg + (size_t)col * Cq, kk, hi);
        v16bf bfm = load_frag_b(Wf + (size_t)col * Cq, kk, hi);
        accg[t] = wmma_bf16(a, bgm, accg[t]);
        accf[t] = wmma_bf16(a, bfm, accf[t]);
      }
    }
#pragma unroll
    for (int t = 0; t < 4; ++t) {
      const int col = (nc * 4 + t) * 16 + l15;
      const float biasg = bg[col];
      const float biasf = bf[col];
#pragma unroll
      for (int j = 0; j < 8; ++j) {
        const int r = r0 + j + hi * 8;
        const int bb = r >> 15;
        const int n = r & (Nq - 1);
        const int m = ix[n];
        const size_t seg = ((size_t)bb * Mq + m) * Cq + col;
        const float e = __expf(accg[t][j] + biasg);
        const float w = e / denom[seg];
        atomicAdd(yagg + seg, (accf[t][j] + biasf) * w);
      }
    }
  }
}

// ---- Pass 3: yh = yagg * Wh^T + bh (2048 aggregated rows) -------------------
__global__ __launch_bounds__(256) void pass3_h(
    const float* __restrict__ yagg, const __bf16* __restrict__ Wh,
    const float* __restrict__ bh, float* __restrict__ yh) {
  const int lane = threadIdx.x & 31;
  const int wave = threadIdx.x >> 5;
  const int l15 = lane & 15, hi = lane >> 4;
  const int r0 = (blockIdx.x * 8 + wave) * 16;  // rows over B*M = 2048
  const float* arow = yagg + (size_t)(r0 + l15) * Cq;

  for (int nc = 0; nc < 8; ++nc) {
    v8f acc[4] = {};
    for (int kk = 0; kk < Cq; kk += 32) {
      v16bf a = load_frag_a_f32(arow, kk, hi);
#pragma unroll
      for (int t = 0; t < 4; ++t) {
        const int col = (nc * 4 + t) * 16 + l15;
        v16bf b = load_frag_b(Wh + (size_t)col * Cq, kk, hi);
        acc[t] = wmma_bf16(a, b, acc[t]);
      }
    }
#pragma unroll
    for (int t = 0; t < 4; ++t) {
      const int col = (nc * 4 + t) * 16 + l15;
      const float bias = bh[col];
#pragma unroll
      for (int j = 0; j < 8; ++j) {
        const int row = r0 + j + hi * 8;
        yh[(size_t)row * Cq + col] = acc[t][j] + bias;
      }
    }
  }
}

// ---- Gather: out[b,n,:] = yh[b, ix[n], :] (float4 vectorized) ---------------
__global__ void gather_expand(const float* __restrict__ yh,
                              const int* __restrict__ ix,
                              float4* __restrict__ out) {
  const int C4 = Cq / 4;
  const size_t total = (size_t)Bq * Nq * C4;
  const size_t stride = (size_t)gridDim.x * blockDim.x;
  for (size_t i = (size_t)blockIdx.x * blockDim.x + threadIdx.x; i < total;
       i += stride) {
    const int c4 = (int)(i % C4);
    const size_t rn = i / C4;
    const int n = (int)(rn % Nq);
    const int b = (int)(rn / Nq);
    const int m = ix[n];
    out[i] = reinterpret_cast<const float4*>(yh)[((size_t)b * Mq + m) * C4 + c4];
  }
}

// ---- Host launch ------------------------------------------------------------
extern "C" void kernel_launch(void* const* d_in, const int* in_sizes, int n_in,
                              void* d_out, int out_size, void* d_ws,
                              size_t ws_size, hipStream_t stream) {
  (void)in_sizes; (void)n_in; (void)out_size; (void)ws_size;
  // setup_inputs order: x, ix, Wf, bf, Wg, bg, Wh, bh
  const float* x  = (const float*)d_in[0];
  const int*   ix = (const int*)d_in[1];
  const float* Wf = (const float*)d_in[2];
  const float* bf = (const float*)d_in[3];
  const float* Wg = (const float*)d_in[4];
  const float* bg = (const float*)d_in[5];
  const float* Wh = (const float*)d_in[6];
  const float* bh = (const float*)d_in[7];
  float* out = (float*)d_out;

  // Workspace layout
  const size_t WN   = (size_t)Cq * Cq;            // 262144 weights elems
  const size_t XN   = (size_t)Bq * Nq * Cq;       // 33.5M x elems
  const size_t SEGN = (size_t)Bq * Mq * Cq;       // 1.05M segment elems
  char* ws = (char*)d_ws;
  __bf16* Wf_h = (__bf16*)(ws);
  __bf16* Wg_h = (__bf16*)(ws + WN * 2);
  __bf16* Wh_h = (__bf16*)(ws + WN * 4);
  __bf16* x_h  = (__bf16*)(ws + WN * 6);
  float* denom = (float*)(ws + WN * 6 + XN * 2);
  float* yagg  = (float*)(ws + WN * 6 + XN * 2 + SEGN * 4);
  float* yh    = (float*)(ws + WN * 6 + XN * 2 + SEGN * 8);

  // 1) precision downshift + clear accumulators (denom & yagg are adjacent)
  cvt_f32_to_bf16<<<512, 256, 0, stream>>>(Wf, Wf_h, (int)WN);
  cvt_f32_to_bf16<<<512, 256, 0, stream>>>(Wg, Wg_h, (int)WN);
  cvt_f32_to_bf16<<<512, 256, 0, stream>>>(Wh, Wh_h, (int)WN);
  cvt_f32_to_bf16<<<8192, 256, 0, stream>>>(x, x_h, (int)XN);
  zero_f32<<<2048, 256, 0, stream>>>(denom, (int)(SEGN * 2));

  // 2) g-GEMM + exp + segment-sum
  pass1_g_exp_scatter<<<(Bq * Nq / 16) / 8, 256, 0, stream>>>(x_h, Wg_h, bg, ix,
                                                              denom);
  // 3) g-GEMM (recompute) + f-GEMM + weighted scatter
  pass2_fw_scatter<<<(Bq * Nq / 16) / 8, 256, 0, stream>>>(x_h, Wg_h, Wf_h, bg,
                                                           bf, ix, denom, yagg);
  // 4) h-GEMM on aggregated segments
  pass3_h<<<(Bq * Mq / 16) / 8, 256, 0, stream>>>(yagg, Wh_h, bh, yh);
  // 5) gather back to [B, N, C]
  gather_expand<<<8192, 256, 0, stream>>>(yh, ix, (float4*)out);
}
